// LLoCaAttention_19782619366074
// MI455X (gfx1250) — compile-verified, matrix operand order checked by reading, not run on hardware
//
#include <hip/hip_runtime.h>
#include <hip/hip_bf16.h>
#include <math.h>

typedef __attribute__((ext_vector_type(16))) _Float16 v16h;
typedef __attribute__((ext_vector_type(8)))  _Float16 v8h;
typedef __attribute__((ext_vector_type(8)))  float    v8f;
typedef __attribute__((ext_vector_type(4)))  float    v4f;

#define NB   256
#define NH   8
#define NN   128
#define NC   64
#define NSC  16   // scalar channels
#define NVV  12   // four-vectors per feature

// LDS strides (chosen for conflict-free WMMA fragment loads)
#define QK_STR 72    // halfs   (64 + 8 pad)
#define VT_STR 136   // halfs   (128 + 8 pad)
#define P_STR  136   // halfs
#define S_STR  132   // floats  (128 + 4 pad)
#define O_STR  68    // floats  (64 + 4 pad)

#define LDS_FL_BYTES   (NN * 16 * 4)             // frames: 8192
#define LDS_QG_BYTES   (NN * QK_STR * 2)         // 18432
#define LDS_KG_BYTES   (NN * QK_STR * 2)         // 18432
#define LDS_VT_BYTES   (NC * VT_STR * 2)         // 17408
#define LDS_S_BYTES    (NN * S_STR * 4)          // 67584
#define LDS_RS_BYTES   (NN * 4)                  // 512
#define LDS_TOTAL      (LDS_FL_BYTES + LDS_QG_BYTES + LDS_KG_BYTES + \
                        LDS_VT_BYTES + LDS_S_BYTES + LDS_RS_BYTES)   // 130560 B

__device__ __forceinline__ v16h load_frag(const _Float16* row, int c0) {
    v8h lo = *(const v8h*)(row + c0);
    v8h hi = *(const v8h*)(row + c0 + 16);
    return __builtin_shufflevector(lo, hi, 0,1,2,3,4,5,6,7,8,9,10,11,12,13,14,15);
}

__global__ __launch_bounds__(256)
void lloca_attn_kernel(const float* __restrict__ q,
                       const float* __restrict__ k,
                       const float* __restrict__ v,
                       const float* __restrict__ frames,
                       float* __restrict__ out)
{
    extern __shared__ char smem_raw[];
    float*    Fl = (float*)smem_raw;                       // [128][16] frames (row-major 4x4)
    _Float16* qg = (_Float16*)(smem_raw + LDS_FL_BYTES);   // [128][QK_STR] f16 (pre-scaled 1/C)
    _Float16* kg = qg + NN * QK_STR;                       // [128][QK_STR] f16
    _Float16* vt = kg + NN * QK_STR;                       // [64][VT_STR]  f16 (v_g transposed)
    float*    Sl = (float*)((char*)vt + LDS_VT_BYTES);     // [128][S_STR]  f32 scores
    float*    rs = Sl + NN * S_STR;                        // [128] 1/rowsum
    _Float16* Pl = qg;                                     // alias: [128][P_STR] f16 probs
    float*    Ol = Sl;                                     // alias: [128][O_STR] f32 out_g

    const int tid = threadIdx.x;
    const int bh  = blockIdx.x;
    const int b   = bh / NH;
    const size_t headOff = (size_t)bh * NN * NC;

    // ---------------- kick off global streams early (global_prefetch_b8) ----------------
    for (int w = tid; w < 3 * NN; w += 256) {
        const int t = w >> 7;
        const int n = w & (NN - 1);
        const float* src = (t == 0 ? q : (t == 1 ? k : v)) + headOff + (size_t)n * NC;
        __builtin_prefetch(src,      0, 1);
        __builtin_prefetch(src + 32, 0, 1);   // second 128B of the 256B row
    }

    // ---------------- stage frames[b] into LDS ----------------
    {
        const v4f* fsrc = (const v4f*)(frames + (size_t)b * NN * 16);
        #pragma unroll
        for (int i = tid; i < NN * 16 / 4; i += 256)
            ((v4f*)Fl)[i] = fsrc[i];
    }
    __syncthreads();

    // ---------------- input reps-transform (fp32 math, f16 store) ----------------
    // q_g, v_g : Minv = F^T          ->  y_u = sum_v F[v][u] * x_v
    // k_g      : g F^T g             ->  y_u = g_u * (F[0][u]x0 - F[1][u]x1 - F[2][u]x2 - F[3][u]x3)
    // The 1/C attention scale is folded into q_g before f16 conversion.
    for (int w = tid; w < 3 * NN; w += 256) {
        const int t = w >> 7;          // 0=q, 1=k, 2=v
        const int n = w & (NN - 1);
        const float* src = (t == 0 ? q : (t == 1 ? k : v)) + headOff + (size_t)n * NC;
        const float* F   = Fl + n * 16;
        const float pre  = (t == 0) ? (1.0f / (float)NC) : 1.0f;

        _Float16 tmp[NC];
        #pragma unroll
        for (int c = 0; c < NSC; ++c) tmp[c] = (_Float16)(src[c] * pre);

        #pragma unroll
        for (int kv = 0; kv < NVV; ++kv) {
            const float x0 = src[NSC + 4*kv + 0];
            const float x1 = src[NSC + 4*kv + 1];
            const float x2 = src[NSC + 4*kv + 2];
            const float x3 = src[NSC + 4*kv + 3];
            #pragma unroll
            for (int u = 0; u < 4; ++u) {
                float y;
                if (t == 1) {
                    float s = F[0*4+u]*x0 - F[1*4+u]*x1 - F[2*4+u]*x2 - F[3*4+u]*x3;
                    y = (u == 0) ? s : -s;
                } else {
                    y = (F[0*4+u]*x0 + F[1*4+u]*x1 + F[2*4+u]*x2 + F[3*4+u]*x3) * pre;
                }
                tmp[NSC + 4*kv + u] = (_Float16)y;
            }
        }

        if (t == 0) {
            #pragma unroll
            for (int c = 0; c < NC; ++c) qg[n * QK_STR + c] = tmp[c];
        } else if (t == 1) {
            #pragma unroll
            for (int c = 0; c < NC; ++c) kg[n * QK_STR + c] = tmp[c];
        } else {
            #pragma unroll
            for (int c = 0; c < NC; ++c) vt[c * VT_STR + n] = tmp[c];
        }
    }
    __syncthreads();

    // ---------------- S = (q_g/C) k_g^T  via WMMA f16 -> f32 ----------------
    const int wv   = tid >> 5;          // wave id 0..7 -> row tile
    const int lane = tid & 31;
    const int lr   = lane & 15;         // fragment row/col within tile
    const int hoff = (lane >> 4) << 3;  // K-half offset (0 or 8)
    const int hi8  = (lane >> 4) << 3;  // C/D row offset (0 or 8)

    {
        // Hoist the two A fragments (whole 16x64 q row-tile) into registers.
        const _Float16* arow = qg + (wv * 16 + lr) * QK_STR;
        const v16h a0 = load_frag(arow, 0  + hoff);
        const v16h a1 = load_frag(arow, 32 + hoff);

        #pragma unroll
        for (int ct = 0; ct < 8; ++ct) {
            const _Float16* brow = kg + (ct * 16 + lr) * QK_STR;
            v8f acc = {};
            acc = __builtin_amdgcn_wmma_f32_16x16x32_f16(
                    false, a0, false, load_frag(brow, 0  + hoff), (short)0, acc, false, false);
            acc = __builtin_amdgcn_wmma_f32_16x16x32_f16(
                    false, a1, false, load_frag(brow, 32 + hoff), (short)0, acc, false, false);
            #pragma unroll
            for (int j = 0; j < 8; ++j)
                Sl[(wv * 16 + hi8 + j) * S_STR + ct * 16 + lr] = acc[j];
        }
    }
    __syncthreads();

    // ---------------- row softmax (fp32), store unnormalized exp as f16 P ----------------
    if (tid < NN) {
        const float* srow = Sl + tid * S_STR;
        float mx = -INFINITY;
        #pragma unroll 4
        for (int j = 0; j < NN; ++j) mx = fmaxf(mx, srow[j]);
        float sum = 0.0f;
        _Float16* prow = Pl + tid * P_STR;
        #pragma unroll 4
        for (int j = 0; j < NN; ++j) {
            float e = __expf(srow[j] - mx);
            sum += e;
            prow[j] = (_Float16)e;
        }
        rs[tid] = 1.0f / sum;
    }
    __syncthreads();

    // ---------------- O = P * V  via WMMA (K = 128) ----------------
    {
        // Hoist the four A fragments (whole 16x128 P row-tile) into registers.
        const _Float16* arow = Pl + (wv * 16 + lr) * P_STR;
        const v16h a0 = load_frag(arow, 0   + hoff);
        const v16h a1 = load_frag(arow, 32  + hoff);
        const v16h a2 = load_frag(arow, 64  + hoff);
        const v16h a3 = load_frag(arow, 96  + hoff);

        #pragma unroll
        for (int ct = 0; ct < 4; ++ct) {
            const _Float16* brow = vt + (ct * 16 + lr) * VT_STR;  // row = out channel
            v8f acc = {};
            acc = __builtin_amdgcn_wmma_f32_16x16x32_f16(
                    false, a0, false, load_frag(brow, 0   + hoff), (short)0, acc, false, false);
            acc = __builtin_amdgcn_wmma_f32_16x16x32_f16(
                    false, a1, false, load_frag(brow, 32  + hoff), (short)0, acc, false, false);
            acc = __builtin_amdgcn_wmma_f32_16x16x32_f16(
                    false, a2, false, load_frag(brow, 64  + hoff), (short)0, acc, false, false);
            acc = __builtin_amdgcn_wmma_f32_16x16x32_f16(
                    false, a3, false, load_frag(brow, 96  + hoff), (short)0, acc, false, false);
            #pragma unroll
            for (int j = 0; j < 8; ++j)
                Ol[(wv * 16 + hi8 + j) * O_STR + ct * 16 + lr] = acc[j];
        }
    }
    __syncthreads();

    // ---------------- epilogue: out = frames * o_g, fold 1/rowsum, fp32 store ----------------
    if (tid < NN) {
        const int n = tid;
        const float rinv = rs[n];
        const float* F    = Fl + n * 16;
        const float* orow = Ol + n * O_STR;
        float* dst = out + headOff + (size_t)n * NC;

        float res[NC];
        #pragma unroll
        for (int c = 0; c < NSC; ++c) res[c] = orow[c] * rinv;
        #pragma unroll
        for (int kv = 0; kv < NVV; ++kv) {
            const float x0 = orow[NSC + 4*kv + 0];
            const float x1 = orow[NSC + 4*kv + 1];
            const float x2 = orow[NSC + 4*kv + 2];
            const float x3 = orow[NSC + 4*kv + 3];
            #pragma unroll
            for (int u = 0; u < 4; ++u)
                res[NSC + 4*kv + u] =
                    rinv * (F[u*4+0]*x0 + F[u*4+1]*x1 + F[u*4+2]*x2 + F[u*4+3]*x3);
        }
        #pragma unroll
        for (int c = 0; c < NC; c += 4) {
            v4f o = { res[c], res[c+1], res[c+2], res[c+3] };
            *(v4f*)(dst + c) = o;
        }
    }
}

extern "C" void kernel_launch(void* const* d_in, const int* in_sizes, int n_in,
                              void* d_out, int out_size, void* d_ws, size_t ws_size,
                              hipStream_t stream) {
    const float* q      = (const float*)d_in[0];
    const float* k      = (const float*)d_in[1];
    const float* v      = (const float*)d_in[2];
    const float* frames = (const float*)d_in[3];
    float* outp = (float*)d_out;

    (void)in_sizes; (void)n_in; (void)out_size; (void)d_ws; (void)ws_size;

    hipFuncSetAttribute((const void*)lloca_attn_kernel,
                        hipFuncAttributeMaxDynamicSharedMemorySize, LDS_TOTAL);

    lloca_attn_kernel<<<dim3(NB * NH), dim3(256), LDS_TOTAL, stream>>>(
        q, k, v, frames, outp);
}
